// LocalizationAttention_21758304321632
// MI455X (gfx1250) — matrix operand order repaired and checked by reading, not compiled, not run on hardware
//
#include <hip/hip_runtime.h>
#include <hip/hip_bf16.h>

// ---------------------------------------------------------------------------
// LocalizationAttention for MI455X (gfx1250, wave32, WMMA).
// B=4, L=1024, D=1024, H=16, HD=64, TERM=20.
// Outputs (flat f32): pooled[4*3072] | attn_mean[4*1024*1024] | entropy[1]
// ---------------------------------------------------------------------------

typedef _Float16 v16h __attribute__((ext_vector_type(16)));
typedef _Float16 v8h  __attribute__((ext_vector_type(8)));
typedef float    v8f  __attribute__((ext_vector_type(8)));

#define BATCH 4
#define SEQ   1024
#define DIM   1024
#define NH    16
#define HD    64
#define TERM  20
#define QBLK  16     // q rows per attention workgroup

// workspace layout (bytes)
#define XNH_OFF   ((size_t)0)                       // 4096*1024 fp16 = 8MB
#define QH_OFF    ((size_t)8  * 1024 * 1024)        // [B,H,L,64] fp16
#define KH_OFF    ((size_t)16 * 1024 * 1024)        // [B,H,L,64] fp16
#define VT_OFF    ((size_t)24 * 1024 * 1024)        // [B,H,64,L] fp16 (transposed V)
#define PACC_OFF  ((size_t)32 * 1024 * 1024)        // 12288 f32 pooled accum
#define ENT_OFF   (PACC_OFF + 12288 * sizeof(float))

// d_out layout (floats)
#define OUT_POOLED 0
#define OUT_ATTN   12288
#define OUT_ENT    (12288 + BATCH * SEQ * SEQ)      // 4206592

static __device__ __forceinline__ v8f wmma_f16(v16h a, v16h b, v8f c) {
  return __builtin_amdgcn_wmma_f32_16x16x32_f16(false, a, false, b, (short)0, c,
                                                false, false);
}

static __device__ __forceinline__ v16h pack16(v8h lo, v8h hi) {
  v16h a;
#pragma unroll
  for (int j = 0; j < 8; j++) { a[j] = lo[j]; a[8 + j] = hi[j]; }
  return a;
}

// ---------------------------------------------------------------------------
// kernel 0: zero pooled/entropy accumulators
// ---------------------------------------------------------------------------
__global__ void zero_acc_kernel(float* p, int n) {
  int i = blockIdx.x * 256 + threadIdx.x;
  if (i < n) p[i] = 0.0f;
}

// ---------------------------------------------------------------------------
// kernel 1: LayerNorm -> fp16 xn.  One block per token, 256 threads.
// ---------------------------------------------------------------------------
__global__ void __launch_bounds__(256)
ln_kernel(const float* __restrict__ x, const float* __restrict__ gamma,
          const float* __restrict__ beta, _Float16* __restrict__ xnh) {
  const int token = blockIdx.x;
  const float* xr = x + (size_t)token * DIM;
  float4 v = ((const float4*)xr)[threadIdx.x];        // 4 contiguous floats
  float s  = v.x + v.y + v.z + v.w;
  float ss = v.x * v.x + v.y * v.y + v.z * v.z + v.w * v.w;
  for (int m = 1; m < 32; m <<= 1) {
    s  += __shfl_xor(s, m, 32);
    ss += __shfl_xor(ss, m, 32);
  }
  __shared__ float red[16];
  const int wave = threadIdx.x >> 5, lane = threadIdx.x & 31;
  if (lane == 0) { red[wave] = s; red[8 + wave] = ss; }
  __syncthreads();
  float ts = 0.f, tss = 0.f;
#pragma unroll
  for (int i = 0; i < 8; i++) { ts += red[i]; tss += red[8 + i]; }
  const float mu  = ts * (1.0f / DIM);
  const float var = tss * (1.0f / DIM) - mu * mu;
  const float rst = rsqrtf(var + 1e-5f);
  const int c0 = threadIdx.x * 4;
  _Float16* o = xnh + (size_t)token * DIM + c0;
  const float vv[4] = {v.x, v.y, v.z, v.w};
#pragma unroll
  for (int j = 0; j < 4; j++)
    o[j] = (_Float16)(((vv[j] - mu) * rst) * gamma[c0 + j] + beta[c0 + j]);
}

// ---------------------------------------------------------------------------
// kernel 2: QKV projection GEMM, fp16 WMMA.
// grid = 3 * (4096/128) * (1024/64) blocks, 256 threads (8 waves).
// Block tile 128(M) x 64(N); each wave owns a 16x64 strip (4 WMMA col-tiles).
// W (f32) staged to LDS transposed [n][k] as fp16 each K-step of 32:
// each thread gathers an 8-deep K column (lane-coalesced in global) and
// writes one packed v8h -> single ds_store_b128.
// Q,K stored [B,H,L,64]; V stored transposed [B,H,64,L].
// ---------------------------------------------------------------------------
__global__ void __launch_bounds__(256)
qkv_gemm_kernel(const _Float16* __restrict__ xnh,
                const float* __restrict__ Wq, const float* __restrict__ bq,
                const float* __restrict__ Wk, const float* __restrict__ bk,
                const float* __restrict__ Wv, const float* __restrict__ bv,
                _Float16* __restrict__ Qh, _Float16* __restrict__ Kh,
                _Float16* __restrict__ Vt) {
  const int bid = blockIdx.x;
  const int mat = bid / (32 * 16);
  const int rem = bid % (32 * 16);
  const int m0  = (rem / 16) * 128;
  const int n0  = (rem % 16) * 64;
  const float* W    = (mat == 0) ? Wq : (mat == 1) ? Wk : Wv;
  const float* bias = (mat == 0) ? bq : (mat == 1) ? bk : bv;

  __shared__ _Float16 Bt[64 * 32];     // [n][k] fp16, 4KB

  const int tid  = threadIdx.x;
  const int wave = tid >> 5;
  const int lane = tid & 31;
  const int half = lane >> 4;          // 0 / 1
  const int l15  = lane & 15;
  const int mrow = m0 + wave * 16;

  // W staging assignment: thread owns (n, kc..kc+7)
  const int sn = tid & 63;             // 0..63 col within tile
  const int sk = (tid >> 6) * 8;       // 0,8,16,24 k-chunk

  v8f acc[4] = {};

  for (int kk0 = 0; kk0 < DIM; kk0 += 32) {
    // stage W[kk0..+31][n0..+63] -> Bt[n][k] fp16 (one b128 LDS store/thread)
    {
      const float* wp = W + (size_t)(kk0 + sk) * DIM + n0 + sn;
      v8h pk;
#pragma unroll
      for (int j = 0; j < 8; j++) pk[j] = (_Float16)wp[(size_t)j * DIM];
      *(v8h*)&Bt[sn * 32 + sk] = pk;
      if (kk0 + 32 < DIM)              // prefetch next K-tile of W
        __builtin_prefetch(wp + (size_t)32 * DIM, 0, 1);
    }
    __syncthreads();

    // A tile: xn rows mrow+l15, K = kk0 + {0..15} | {16..31}
    const _Float16* ap = xnh + (size_t)(mrow + l15) * DIM + kk0 + half * 16;
    v16h a = pack16(*(const v8h*)(ap), *(const v8h*)(ap + 8));

#pragma unroll
    for (int t = 0; t < 4; t++) {
      const _Float16* bp = &Bt[(t * 16 + l15) * 32 + half * 16];
      v16h bm = *(const v16h*)bp;
      acc[t] = wmma_f16(a, bm, acc[t]);
    }
    __syncthreads();
  }

  // epilogue: +bias, scatter to head-major layouts
#pragma unroll
  for (int t = 0; t < 4; t++) {
    const int ng = n0 + t * 16 + l15;
    const float bs = bias[ng];
    const int h = ng >> 6, d = ng & 63;
#pragma unroll
    for (int r = 0; r < 8; r++) {
      const int mg = mrow + half * 8 + r;
      const int bb = mg >> 10, l = mg & 1023;
      const _Float16 val = (_Float16)(acc[t][r] + bs);
      if (mat == 2)
        Vt[(((size_t)bb * NH + h) * HD + d) * SEQ + l] = val;
      else if (mat == 0)
        Qh[(((size_t)bb * NH + h) * SEQ + l) * HD + d] = val;
      else
        Kh[(((size_t)bb * NH + h) * SEQ + l) * HD + d] = val;
    }
  }
}

// ---------------------------------------------------------------------------
// kernel 3: fused attention. grid = B * (L/16) = 256 blocks, 256 threads.
// LDS: S[16][1024] f32 scores, Asum[16][1024] f32 head-mean accum,
//      Ph[16][1024] fp16 probabilities (WMMA A operand), Qs[16][64] fp16,
//      red[4*32*8] f32 split-K partials.  ~166KB (WGP has 320KB).
// ---------------------------------------------------------------------------
#define SMEM3 (QBLK * SEQ * 4 /*S*/ + QBLK * SEQ * 4 /*Asum*/ + \
               QBLK * SEQ * 2 /*Ph*/ + QBLK * HD * 2 /*Qs*/ +   \
               4 * 32 * 8 * 4 /*red*/ + 64 /*entred*/)

__global__ void __launch_bounds__(256)
attn_kernel(const _Float16* __restrict__ Qh, const _Float16* __restrict__ Kh,
            const _Float16* __restrict__ Vt, const unsigned char* __restrict__ mask,
            float* __restrict__ pooled_acc, float* __restrict__ ent_acc,
            float* __restrict__ out_attn) {
  extern __shared__ char smem[];
  float*    S    = (float*)smem;                   // 16*1024 f32
  float*    Asum = S + QBLK * SEQ;                 // 16*1024 f32
  _Float16* Ph   = (_Float16*)(Asum + QBLK * SEQ); // 16*1024 fp16
  _Float16* Qs   = Ph + QBLK * SEQ;                // 16*64 fp16
  float*    red  = (float*)(Qs + QBLK * HD);       // 4*32*8 f32
  float*    entred = red + 4 * 32 * 8;             // 8 f32

  const int b  = blockIdx.x >> 6;                  // 256 blocks = 4 * 64
  const int q0 = (blockIdx.x & 63) * QBLK;
  const int tid = threadIdx.x;
  const int wave = tid >> 5, lane = tid & 31;
  const int half = lane >> 4, l15 = lane & 15;

  // sequence length from contiguous-prefix mask
  __shared__ int len_sh;
  if (tid == 0) len_sh = 0;
  __syncthreads();
  {
    int cnt = 0;
    const unsigned char* mb = mask + (size_t)b * SEQ;
    for (int i = tid; i < SEQ; i += 256) cnt += (mb[i] != 0);
    for (int m = 1; m < 32; m <<= 1) cnt += __shfl_xor(cnt, m, 32);
    if (lane == 0) atomicAdd(&len_sh, cnt);
  }
  for (int i = tid; i < QBLK * SEQ; i += 256) Asum[i] = 0.0f;
  __syncthreads();
  const int   len       = len_sh;
  const int   term      = (len < TERM) ? len : TERM;
  const float inv_term  = 1.0f / (float)term;
  const float inv_scale = 0.35355339059327373f;    // HD^-0.25 = 1/(2*sqrt(2))

  const int srow = tid >> 4;      // softmax: 16 threads per row (same wave)
  const int ssub = tid & 15;
  float ent_local = 0.0f;

  for (int h = 0; h < NH; ++h) {
    const _Float16* Qb = Qh + (((size_t)b * NH + h) * SEQ + q0) * HD;
    const _Float16* Kb = Kh + (((size_t)b * NH + h) * SEQ) * HD;
    const _Float16* Vb = Vt + (((size_t)b * NH + h) * HD) * SEQ;

    for (int i = tid; i < QBLK * HD; i += 256) Qs[i] = Qb[i];
    __syncthreads();

    // ---- scores S = Q Kt * inv_scale + bias, masked ----------------------
    for (int i = 0; i < 8; i++) {              // 64 col-tiles / 8 waves
      const int ct = wave + 8 * i;
      v8f acc = {};
#pragma unroll
      for (int hs = 0; hs < 2; hs++) {         // HD=64 -> two K=32 steps
        const int hd0 = hs * 32;
        const _Float16* ap = Qs + l15 * HD + hd0 + half * 16;
        v16h a = pack16(*(const v8h*)ap, *(const v8h*)(ap + 8));
        const _Float16* bp = Kb + (size_t)(ct * 16 + l15) * HD + hd0 + half * 16;
        v16h bm = *(const v16h*)bp;            // Kt: 16 contiguous hd values
        acc = wmma_f16(a, bm, acc);
      }
      const int  kpos = ct * 16 + l15;         // column is per-lane constant
      const bool inl  = kpos < len;
      float bias;
      if (len >= TERM)
        bias = (kpos < TERM ? 1.0f : 0.0f) +
               ((inl && kpos >= len - TERM) ? 0.8f : 0.0f);
      else
        bias = inl ? 1.8f : 0.0f;
#pragma unroll
      for (int r = 0; r < 8; r++) {
        const float sv = inl ? acc[r] * inv_scale + bias : -10000.0f;
        S[(half * 8 + r) * SEQ + kpos] = sv;
      }
    }
    __syncthreads();

    // ---- softmax + entropy; probabilities -> Ph (fp16) + Asum (f32) ------
    {
      float* Sr = S + srow * SEQ;
      float* Ar = Asum + srow * SEQ;
      _Float16* Pr = Ph + srow * SEQ;
      float mx = -3.0e38f;
      for (int i = ssub; i < SEQ; i += 16) mx = fmaxf(mx, Sr[i]);
      for (int m = 1; m < 16; m <<= 1) mx = fmaxf(mx, __shfl_xor(mx, m, 16));
      float sum = 0.0f;
      for (int i = ssub; i < SEQ; i += 16) {
        const float e = __expf(Sr[i] - mx);
        Sr[i] = e;
        sum += e;
      }
      for (int m = 1; m < 16; m <<= 1) sum += __shfl_xor(sum, m, 16);
      const float inv = 1.0f / sum;
      for (int i = ssub; i < SEQ; i += 16) {
        const float p = Sr[i] * inv;
        Pr[i] = (_Float16)p;
        Ar[i] += p;
        ent_local += p * __logf(p + 1e-10f);
      }
    }
    __syncthreads();

    // ---- context O = P @ V (split-K across wave pairs) -------------------
    {
      const int ct2   = wave & 3;              // hd col-tile 0..3
      const int khalf = wave >> 2;             // 0: k<512, 1: k>=512
      v8f o = {};
      const int kbeg = khalf * 512;
      for (int k0 = kbeg; k0 < kbeg + 512; k0 += 32) {
        const _Float16* ap = Ph + l15 * SEQ + k0 + half * 16;
        v16h a = pack16(*(const v8h*)ap, *(const v8h*)(ap + 8));
        const _Float16* bp = Vb + (size_t)(ct2 * 16 + l15) * SEQ + k0 + half * 16;
        v16h bm = *(const v16h*)bp;            // Vt: 16 contiguous tokens
        o = wmma_f16(a, bm, o);
      }
      if (khalf == 1) {
        float* rp = red + (ct2 * 32 + lane) * 8;
#pragma unroll
        for (int r = 0; r < 8; r++) rp[r] = o[r];
      }
      __syncthreads();
      if (khalf == 0) {
        const float* rp = red + (ct2 * 32 + lane) * 8;
        const int dcol = h * HD + ct2 * 16 + l15;
        float snv = 0.f, scv = 0.f, sgv = 0.f;
#pragma unroll
        for (int r = 0; r < 8; r++) {
          const int q = q0 + half * 8 + r;
          const float v = o[r] + rp[r];
          if (q < term) snv += v;
          if (q >= len - term && q < len) scv += v;
          sgv += v;
        }
        atomicAdd(&pooled_acc[b * 3072 + dcol],        snv * inv_term);
        atomicAdd(&pooled_acc[b * 3072 + 1024 + dcol], scv * inv_term);
        atomicAdd(&pooled_acc[b * 3072 + 2048 + dcol], sgv * (1.0f / SEQ));
      }
    }
    __syncthreads();   // protect S/Ph/Qs/red for next head
  }

  // ---- write attn.mean over heads ----------------------------------------
  float* ao = out_attn + ((size_t)b * SEQ + q0) * SEQ;
  for (int i = tid; i < QBLK * SEQ; i += 256) ao[i] = Asum[i] * (1.0f / NH);

  // ---- entropy partial -> global ------------------------------------------
  for (int m = 1; m < 32; m <<= 1) ent_local += __shfl_xor(ent_local, m, 32);
  if (lane == 0) entred[wave] = ent_local;
  __syncthreads();
  if (tid == 0) {
    float e = 0.f;
#pragma unroll
    for (int i = 0; i < 8; i++) e += entred[i];
    atomicAdd(ent_acc, e);
  }
}

// ---------------------------------------------------------------------------
// kernel 4: finalize pooled + entropy into d_out
// ---------------------------------------------------------------------------
__global__ void finalize_kernel(const float* __restrict__ pacc,
                                const float* __restrict__ ent,
                                float* __restrict__ out) {
  const int i = blockIdx.x * 256 + threadIdx.x;
  if (i < 12288) out[OUT_POOLED + i] = pacc[i];
  if (i == 0)
    out[OUT_ENT] = -ent[0] * (1.0f / (float)(BATCH * NH * SEQ));
}

// ---------------------------------------------------------------------------
extern "C" void kernel_launch(void* const* d_in, const int* in_sizes, int n_in,
                              void* d_out, int out_size, void* d_ws,
                              size_t ws_size, hipStream_t stream) {
  const float* x     = (const float*)d_in[0];
  const unsigned char* mask = (const unsigned char*)d_in[1];
  const float* Wq    = (const float*)d_in[2];
  const float* bq    = (const float*)d_in[3];
  const float* Wk    = (const float*)d_in[4];
  const float* bk    = (const float*)d_in[5];
  const float* Wv    = (const float*)d_in[6];
  const float* bv    = (const float*)d_in[7];
  const float* gamma = (const float*)d_in[8];
  const float* beta  = (const float*)d_in[9];
  float* out = (float*)d_out;

  char* ws = (char*)d_ws;
  _Float16* xnh = (_Float16*)(ws + XNH_OFF);
  _Float16* Qh  = (_Float16*)(ws + QH_OFF);
  _Float16* Kh  = (_Float16*)(ws + KH_OFF);
  _Float16* Vt  = (_Float16*)(ws + VT_OFF);
  float*    pacc = (float*)(ws + PACC_OFF);
  float*    ent  = (float*)(ws + ENT_OFF);

  // 0) zero accumulators (12288 pooled + 1 entropy)
  zero_acc_kernel<<<(12289 + 255) / 256, 256, 0, stream>>>(pacc, 12289);

  // 1) LayerNorm -> fp16
  ln_kernel<<<BATCH * SEQ, 256, 0, stream>>>(x, gamma, beta, xnh);

  // 2) QKV projections (WMMA)
  qkv_gemm_kernel<<<3 * 32 * 16, 256, 0, stream>>>(xnh, Wq, bq, Wk, bk, Wv, bv,
                                                   Qh, Kh, Vt);

  // 3) fused attention (WMMA + LDS softmax), writes attn_mean region
  attn_kernel<<<BATCH * (SEQ / QBLK), 256, SMEM3, stream>>>(
      Qh, Kh, Vt, mask, pacc, ent, out + OUT_ATTN);

  // 4) finalize pooled + entropy
  finalize_kernel<<<(12288 + 255) / 256, 256, 0, stream>>>(pacc, ent, out);
}
